// GeneralSAETopK_2448131359470
// MI455X (gfx1250) — matrix operand order repaired and checked by reading, not compiled, not run on hardware
//
#include <hip/hip_runtime.h>

// ---------------------------------------------------------------------------
// Problem constants (from reference setup_inputs)
// ---------------------------------------------------------------------------
#define BATCH 8192
#define FEAT  24576
#define KDIM  768
#define TOPK  32

// GEMM tiling
#define BM 128
#define BN 256
#define BK 32
#define LDSS 80  // LDS row stride in bytes (32 bf16 = 64B data, padded to 80)

typedef __attribute__((ext_vector_type(16))) __bf16 v16bf;
typedef __attribute__((ext_vector_type(8)))  float  v8f;
typedef __attribute__((ext_vector_type(4)))  float  f32x4;  // native vector for NT builtins

union Frag {
    v16bf b;
    uint4 q[2];
};

__device__ __forceinline__ unsigned int f2bf(float f) {
    unsigned int u = __float_as_uint(f);
    unsigned int r = u + 0x7FFFu + ((u >> 16) & 1u);   // round-to-nearest-even
    return r >> 16;
}
__device__ __forceinline__ unsigned int pk_bf16(float lo, float hi) {
    return (f2bf(lo) & 0xFFFFu) | (f2bf(hi) << 16);
}

// ---------------------------------------------------------------------------
// Kernel 1a: per-column sum of squares of D (partials, no float atomics ->
// bitwise deterministic across graph replays).
// D is [F, K] row-major; norm is over axis 0 (length F) per column m.
// 96 blocks, each handles 256 rows; thread t owns columns {t, t+256, t+512}.
// ---------------------------------------------------------------------------
__global__ __launch_bounds__(256) void sae_norms_partial(
    const float* __restrict__ D, float* __restrict__ partial)
{
    const int t  = threadIdx.x;
    const int r0 = blockIdx.x * 256;
    float a0 = 0.f, a1 = 0.f, a2 = 0.f;
    for (int r = 0; r < 256; ++r) {
        const float* row = D + (size_t)(r0 + r) * KDIM;
        float x0 = row[t], x1 = row[t + 256], x2 = row[t + 512];
        a0 += x0 * x0; a1 += x1 * x1; a2 += x2 * x2;
    }
    float* p = partial + (size_t)blockIdx.x * KDIM;
    p[t] = a0; p[t + 256] = a1; p[t + 512] = a2;
}

// Kernel 1b: reduce 96 partials -> inv_norm[m] = rsqrt(sum)
__global__ __launch_bounds__(256) void sae_norms_reduce(
    const float* __restrict__ partial, float* __restrict__ inv_norm)
{
    const int m = blockIdx.x * 256 + threadIdx.x;   // grid = 3
    float s = 0.f;
    for (int b = 0; b < 96; ++b) s += partial[(size_t)b * KDIM + m];
    inv_norm[m] = rsqrtf(s);
}

// ---------------------------------------------------------------------------
// Kernel 2: encoder GEMM  S = (X - pre_bias) @ W^T + (b_enc + latent_bias)
// bf16 WMMA, f32 accumulate. Block 256 thr (8 waves), tile 128x256x32.
// Wave tile 64x64 -> 16 x v_wmma_f32_16x16x32_bf16 per K-step.
// Epilogue stores are non-temporal: the 805MB s_pre stream must not evict
// the L2-resident operands (X+W+D = 176MB fits the 192MB L2).
// ---------------------------------------------------------------------------
__global__ __launch_bounds__(256) void sae_enc_gemm(
    const float* __restrict__ X, const float* __restrict__ W,
    const float* __restrict__ b_enc, const float* __restrict__ lat_b,
    const float* __restrict__ pre_bias, float* __restrict__ S)
{
    __shared__ uint4 smem[(BM * LDSS + BN * LDSS) / 16];
    unsigned char* Xs = (unsigned char*)smem;              // 128 rows x 80B
    unsigned char* Wl = (unsigned char*)smem + BM * LDSS;  // 256 rows x 80B

    const int tid   = threadIdx.x;
    const int lane  = tid & 31;
    const int wid   = tid >> 5;
    const int hi    = (lane >> 4) & 1;
    const int lrow  = lane & 15;
    const int bm0   = blockIdx.y << 7;            // * BM
    const int bn0   = blockIdx.x << 8;            // * BN
    const int waveM = (wid >> 2) << 6;            // 0 / 64
    const int waveN = (wid & 3) << 6;             // 0,64,128,192

    v8f acc[4][4];
    #pragma unroll
    for (int i = 0; i < 4; ++i)
        #pragma unroll
        for (int j = 0; j < 4; ++j)
            acc[i][j] = (v8f){0.f,0.f,0.f,0.f,0.f,0.f,0.f,0.f};

    // staging assignment: X tile: thread covers half-row (16 floats);
    // W tile: thread covers one full 32-float row.
    const int   xr  = tid >> 1;
    const int   xcb = (tid & 1) << 4;
    const float* xsrc = X + (size_t)(bm0 + xr) * KDIM + xcb;
    const float* wsrc = W + (size_t)(bn0 + tid) * KDIM;

    for (int k0 = 0; k0 < KDIM; k0 += BK) {
        { // stage X tile (subtract pre_bias, cvt bf16)
            const float* s  = xsrc + k0;
            const float* pb = pre_bias + k0 + xcb;
            float4 f[4], p[4];
            #pragma unroll
            for (int q = 0; q < 4; ++q) {
                f[q] = *(const float4*)(s  + 4 * q);
                p[q] = *(const float4*)(pb + 4 * q);
            }
            unsigned int u[8];
            #pragma unroll
            for (int q = 0; q < 4; ++q) {
                u[2*q]   = pk_bf16(f[q].x - p[q].x, f[q].y - p[q].y);
                u[2*q+1] = pk_bf16(f[q].z - p[q].z, f[q].w - p[q].w);
            }
            uint4* dst = (uint4*)(Xs + xr * LDSS + xcb * 2);
            dst[0] = make_uint4(u[0], u[1], u[2], u[3]);
            dst[1] = make_uint4(u[4], u[5], u[6], u[7]);
        }
        { // stage W tile (cvt bf16)
            const float* s = wsrc + k0;
            float4 f[8];
            #pragma unroll
            for (int q = 0; q < 8; ++q) f[q] = *(const float4*)(s + 4 * q);
            unsigned int u[16];
            #pragma unroll
            for (int q = 0; q < 8; ++q) {
                u[2*q]   = pk_bf16(f[q].x, f[q].y);
                u[2*q+1] = pk_bf16(f[q].z, f[q].w);
            }
            uint4* dst = (uint4*)(Wl + tid * LDSS);
            #pragma unroll
            for (int q = 0; q < 4; ++q)
                dst[q] = make_uint4(u[4*q], u[4*q+1], u[4*q+2], u[4*q+3]);
        }
        __syncthreads();

        // A frags: lane holds row M=lrow, K chunks {8*hi..8*hi+7, +16}
        Frag a[4], b[4];
        #pragma unroll
        for (int mi = 0; mi < 4; ++mi) {
            const unsigned char* p = Xs + (waveM + mi * 16 + lrow) * LDSS + (hi ? 16 : 0);
            a[mi].q[0] = *(const uint4*)p;
            a[mi].q[1] = *(const uint4*)(p + 32);
        }
        // B frags: lane holds feature col lrow, K chunks {8*hi..8*hi+7, +16}
        #pragma unroll
        for (int ni = 0; ni < 4; ++ni) {
            const unsigned char* p = Wl + (waveN + ni * 16 + lrow) * LDSS + (hi ? 16 : 0);
            b[ni].q[0] = *(const uint4*)p;
            b[ni].q[1] = *(const uint4*)(p + 32);
        }
        #pragma unroll
        for (int mi = 0; mi < 4; ++mi)
            #pragma unroll
            for (int ni = 0; ni < 4; ++ni)
                acc[mi][ni] = __builtin_amdgcn_wmma_f32_16x16x32_bf16(
                    false, a[mi].b, false, b[ni].b, (short)0, acc[mi][ni],
                    false, false);

        __syncthreads();
    }

    // store C + bias (non-temporal); C/D layout: VGPR v = row (v + 8*hi), col = lrow
    #pragma unroll
    for (int ni = 0; ni < 4; ++ni) {
        const int col  = bn0 + waveN + ni * 16 + lrow;
        const float bias = b_enc[col] + lat_b[col];
        #pragma unroll
        for (int mi = 0; mi < 4; ++mi) {
            const int rbase = bm0 + waveM + mi * 16 + (hi ? 8 : 0);
            v8f c = acc[mi][ni];
            #pragma unroll
            for (int v = 0; v < 8; ++v)
                __builtin_nontemporal_store(c[v] + bias,
                    S + (size_t)(rbase + v) * FEAT + col);
        }
    }
}

// ---------------------------------------------------------------------------
// Kernel 3: per-row exact top-32 (iterative block argmax), rewrite S row
// sparsified (relu on kept values), emit compact (idx, val) lists.
// 1024 threads/row; each thread owns a contiguous 24-element chunk so row
// I/O is 6 x b128 non-temporal loads + 6 x b128 non-temporal stores.
// ---------------------------------------------------------------------------
__global__ __launch_bounds__(1024) void sae_topk(
    float* __restrict__ S, int* __restrict__ topi, float* __restrict__ topv)
{
    const int row  = blockIdx.x;
    const int tid  = threadIdx.x;
    const int lane = tid & 31;
    const int wid  = tid >> 5;
    float* Srow = S + (size_t)row * FEAT;
    const int c0 = tid * 24;                 // this thread's first column

    float v[24];
    #pragma unroll
    for (int q = 0; q < 6; ++q) {
        f32x4 f = __builtin_nontemporal_load((const f32x4*)(Srow + c0) + q);
        v[4*q + 0] = f[0]; v[4*q + 1] = f[1]; v[4*q + 2] = f[2]; v[4*q + 3] = f[3];
    }

    __shared__ float sv[32];
    __shared__ int   sc[32];
    __shared__ int   winc;
    unsigned int taken = 0u;

    for (int r = 0; r < TOPK; ++r) {
        float bv = -3.4e38f;
        int   bc = 0x7fffffff;
        #pragma unroll
        for (int i = 0; i < 24; ++i) {
            bool ok = ((taken >> i) & 1u) == 0u;
            if (ok && v[i] > bv) { bv = v[i]; bc = c0 + i; }
        }
        #pragma unroll
        for (int off = 16; off > 0; off >>= 1) {
            float ov = __shfl_xor(bv, off, 32);
            int   oc = __shfl_xor(bc, off, 32);
            if (ov > bv) { bv = ov; bc = oc; }
        }
        if (lane == 0) { sv[wid] = bv; sc[wid] = bc; }
        __syncthreads();
        if (wid == 0) {
            float bv2 = sv[lane];
            int   bc2 = sc[lane];
            #pragma unroll
            for (int off = 16; off > 0; off >>= 1) {
                float ov = __shfl_xor(bv2, off, 32);
                int   oc = __shfl_xor(bc2, off, 32);
                if (ov > bv2) { bv2 = ov; bc2 = oc; }
            }
            if (lane == 0) {
                winc = bc2;
                topi[row * TOPK + r] = bc2;
                topv[row * TOPK + r] = fmaxf(bv2, 0.f);   // relu(vals)
            }
        }
        __syncthreads();
        const unsigned int d = (unsigned int)(winc - c0);
        if (d < 24u) taken |= 1u << d;
    }

    // rewrite sparsified row: relu(val) at kept positions, 0 elsewhere
    #pragma unroll
    for (int q = 0; q < 6; ++q) {
        f32x4 o;
        o[0] = ((taken >> (4*q + 0)) & 1u) ? fmaxf(v[4*q + 0], 0.f) : 0.f;
        o[1] = ((taken >> (4*q + 1)) & 1u) ? fmaxf(v[4*q + 1], 0.f) : 0.f;
        o[2] = ((taken >> (4*q + 2)) & 1u) ? fmaxf(v[4*q + 2], 0.f) : 0.f;
        o[3] = ((taken >> (4*q + 3)) & 1u) ? fmaxf(v[4*q + 3], 0.f) : 0.f;
        __builtin_nontemporal_store(o, (f32x4*)(Srow + c0) + q);
    }
}

// ---------------------------------------------------------------------------
// Kernel 4: sparse decoder  X_recon[b,m] = (sum_j val_j * D[idx_j, m]) *
//           inv_norm[m] + pre_bias[m].  One block (256 thr) per batch row.
//           D rows stay L2-resident; output stream is non-temporal.
// ---------------------------------------------------------------------------
__global__ __launch_bounds__(256) void sae_decode(
    const float* __restrict__ D, const float* __restrict__ pre_bias,
    const int* __restrict__ topi, const float* __restrict__ topv,
    const float* __restrict__ inv_norm, float* __restrict__ Xr)
{
    const int row = blockIdx.x;
    const int t   = threadIdx.x;
    __shared__ int   si[TOPK];
    __shared__ float svv[TOPK];
    if (t < TOPK) {
        si[t]  = topi[row * TOPK + t];
        svv[t] = topv[row * TOPK + t];
    }
    __syncthreads();

    float a0 = 0.f, a1 = 0.f, a2 = 0.f;
    #pragma unroll 4
    for (int j = 0; j < TOPK; ++j) {
        const float* drow = D + (size_t)si[j] * KDIM;
        const float  vv = svv[j];
        a0 += vv * drow[t];
        a1 += vv * drow[t + 256];
        a2 += vv * drow[t + 512];
    }
    float* out = Xr + (size_t)row * KDIM;
    __builtin_nontemporal_store(a0 * inv_norm[t]       + pre_bias[t],       out + t);
    __builtin_nontemporal_store(a1 * inv_norm[t + 256] + pre_bias[t + 256], out + t + 256);
    __builtin_nontemporal_store(a2 * inv_norm[t + 512] + pre_bias[t + 512], out + t + 512);
}

// ---------------------------------------------------------------------------
// Host launch
// ---------------------------------------------------------------------------
extern "C" void kernel_launch(void* const* d_in, const int* in_sizes, int n_in,
                              void* d_out, int out_size, void* d_ws, size_t ws_size,
                              hipStream_t stream) {
    (void)in_sizes; (void)n_in; (void)out_size; (void)ws_size;
    const float* X   = (const float*)d_in[0];   // [B, K]
    const float* W   = (const float*)d_in[1];   // [F, K]
    const float* be  = (const float*)d_in[2];   // [F]
    const float* lb  = (const float*)d_in[3];   // [F]
    const float* D   = (const float*)d_in[4];   // [F, K]
    const float* pb  = (const float*)d_in[5];   // [K]
    // d_in[6] = k (== 32, compiled in)

    float* S  = (float*)d_out;                              // [B, F]
    float* Xr = (float*)d_out + (size_t)BATCH * FEAT;       // [B, K]

    char*  ws       = (char*)d_ws;
    float* inv_norm = (float*)ws;                                       // 768 f
    float* partial  = (float*)(ws + 4096);                              // 96*768 f
    int*   topi     = (int*)  (ws + 4096 + (size_t)96 * KDIM * 4);      // B*32 i
    float* topv     = (float*)(ws + 4096 + (size_t)96 * KDIM * 4
                                       + (size_t)BATCH * TOPK * 4);     // B*32 f

    sae_norms_partial<<<dim3(96),            dim3(256),  0, stream>>>(D, partial);
    sae_norms_reduce <<<dim3(3),             dim3(256),  0, stream>>>(partial, inv_norm);
    sae_enc_gemm     <<<dim3(FEAT / BN, BATCH / BM), dim3(256), 0, stream>>>(X, W, be, lb, pb, S);
    sae_topk         <<<dim3(BATCH),         dim3(1024), 0, stream>>>(S, topi, topv);
    sae_decode       <<<dim3(BATCH),         dim3(256),  0, stream>>>(D, pb, topi, topv, inv_norm, Xr);
}
